// CoupledMambaFusion_17239998726200
// MI455X (gfx1250) — matrix operand order repaired and checked by reading, not compiled, not run on hardware
//
#include <hip/hip_runtime.h>
#include <hip/hip_bf16.h>

// ---------------------------------------------------------------------------
// CoupledMambaFusion for MI455X (gfx1250).
//   - All dense projections via V_WMMA_F32_16X16X32_BF16 (bf16 in, f32 acc),
//     2x2 register-tiled: 4 WMMAs per 4 fragment loads per K-step.
//   - wave32-aware reductions (__shfl_xor width 32).
//   - Sequential selective scan: one lane per (batch, channel), h[16] state in
//     registers, 2048 steps.
// Workspace layout (floats), M = 8192 rows:
//   vb,ab,tb,vin,ain,tin,xn : 7 * M*256
//   xz : M*1024   uc : M*512   dbc : M*48   delta/y : M*512
//   total = M*3888 floats ~= 121.5 MiB
// ---------------------------------------------------------------------------

#define D_MODEL 256
#define D_INNER 512
#define D_STATE 16
#define D_CONV  4
#define D_TR    16
#define N_SEQ   2048
#define BATCH   4
#define M_ROWS  (BATCH * N_SEQ)   // 8192
#define NLAYER  2
#define EPSV    1e-5f

typedef __bf16 bf16;
typedef __attribute__((ext_vector_type(16))) __bf16 v16bf;
typedef __attribute__((ext_vector_type(8)))  float  v8f;

// A-fragment: lane holds row (lane&15), K elements [hf*8, hf*8+8) then
// [16+hf*8, 16+hf*8+8).  p must already point at rowbase + hf*8.
__device__ __forceinline__ v16bf load_afrag(const float* __restrict__ p)
{
    v16bf f;
#pragma unroll
    for (int i = 0; i < 8; ++i) {
        f[i]     = (bf16)p[i];
        f[8 + i] = (bf16)p[16 + i];
    }
    return f;
}

// B-fragment: lane holds column (lane&15), K elements [hf*8, hf*8+8) then
// [16+hf*8, 16+hf*8+8).  p must already point at colbase + hf*8.
__device__ __forceinline__ v16bf load_bfrag(const float* __restrict__ p)
{
    v16bf f;
#pragma unroll
    for (int i = 0; i < 8; ++i) {
        f[i]     = (bf16)p[i];
        f[8 + i] = (bf16)p[16 + i];
    }
    return f;
}

// ---------------------------------------------------------------------------
// Generic GEMM: Y[m,n] = alpha * sum_k X[m,k] * W[n,k]  (+ bias[n]) (+ resid[m,n])
// W is row-major [N, ldw].  K must be a multiple of 32 (all call sites:
// 256 / 512).  Each wave computes a 32x32 output block (2x2 tiles of
// 16x16); a 256-thread block (8 waves, 2x4) computes 64x128.
// ---------------------------------------------------------------------------
__global__ __launch_bounds__(256) void k_gemm_wmma(
    const float* __restrict__ X, const float* __restrict__ W,
    const float* __restrict__ bias, const float* resid,
    float* Y, int M, int N, int K, int ldw,
    const float* __restrict__ alpha_p, int alpha_idx)
{
    const int wave = threadIdx.x >> 5;
    const int lane = threadIdx.x & 31;
    const int wm = wave & 1;          // 0..1 -> 32-row strips
    const int wn = wave >> 1;         // 0..3 -> 32-col strips
    const int row0 = blockIdx.y * 64 + wm * 32;   // M == multiple of 64
    const int col0 = blockIdx.x * 128 + wn * 32;
    if (row0 >= M || col0 >= N) return;   // wave-uniform: EXEC all-1 for WMMA
    const bool has1 = (col0 + 16) < N;    // second column tile valid?

    const int hf  = lane >> 4;
    const int l16 = lane & 15;

    const float* ap0 = X + (size_t)(row0 + l16) * K + hf * 8;
    const float* ap1 = ap0 + (size_t)16 * K;                 // rows +16
    const float* bp0 = W + (size_t)(col0 + l16) * ldw + hf * 8;
    const float* bp1 = bp0 + (size_t)16 * ldw;               // cols +16

    v8f acc00 = {0.f,0.f,0.f,0.f,0.f,0.f,0.f,0.f};
    v8f acc01 = acc00, acc10 = acc00, acc11 = acc00;

    for (int k = 0; k < K; k += 32) {
        const v16bf a0 = load_afrag(ap0 + k);
        const v16bf a1 = load_afrag(ap1 + k);
        const v16bf b0 = load_bfrag(bp0 + k);
        if (k + 128 < K) {
            __builtin_prefetch(ap0 + k + 128, 0, 0);   // global_prefetch_b8
            __builtin_prefetch(ap1 + k + 128, 0, 0);
            __builtin_prefetch(bp0 + k + 128, 0, 0);
        }
        acc00 = __builtin_amdgcn_wmma_f32_16x16x32_bf16(
                    false, a0, false, b0, (short)0, acc00, false, false);
        acc10 = __builtin_amdgcn_wmma_f32_16x16x32_bf16(
                    false, a1, false, b0, (short)0, acc10, false, false);
        if (has1) {
            const v16bf b1 = load_bfrag(bp1 + k);
            acc01 = __builtin_amdgcn_wmma_f32_16x16x32_bf16(
                        false, a0, false, b1, (short)0, acc01, false, false);
            acc11 = __builtin_amdgcn_wmma_f32_16x16x32_bf16(
                        false, a1, false, b1, (short)0, acc11, false, false);
        }
    }

    const float alpha = alpha_p ? alpha_p[alpha_idx] : 1.0f;

    // C/D layout: lane L, vgpr i -> row (L>=16 ? 8 : 0)+i, col L&15.
    auto store_tile = [&](const v8f& acc, int rb, int cb) {
        const int col   = cb + l16;
        const int rbase = rb + hf * 8;
        const float badd = bias ? bias[col] : 0.0f;
#pragma unroll
        for (int i = 0; i < 8; ++i) {
            float v = alpha * acc[i] + badd;
            if (resid) v += resid[(size_t)(rbase + i) * N + col];
            Y[(size_t)(rbase + i) * N + col] = v;
        }
    };

    store_tile(acc00, row0,      col0);
    store_tile(acc10, row0 + 16, col0);
    if (has1) {
        store_tile(acc01, row0,      col0 + 16);
        store_tile(acc11, row0 + 16, col0 + 16);
    }
}

// ---------------------------------------------------------------------------
// LayerNorm over D_MODEL=256: one wave32 per row, 8 elements per lane.
// Safe for in-place (y == x).
// ---------------------------------------------------------------------------
__global__ __launch_bounds__(256) void k_layernorm(
    const float* __restrict__ x, const float* __restrict__ w,
    const float* __restrict__ b, float* y, int M)
{
    const int wave = threadIdx.x >> 5;
    const int lane = threadIdx.x & 31;
    const int row  = blockIdx.x * 8 + wave;
    if (row >= M) return;
    const float* xr = x + (size_t)row * D_MODEL;
    float v[8];
    float s = 0.f;
#pragma unroll
    for (int j = 0; j < 8; ++j) { v[j] = xr[lane + j * 32]; s += v[j]; }
#pragma unroll
    for (int off = 16; off > 0; off >>= 1) s += __shfl_xor(s, off, 32);
    const float mean = s * (1.0f / D_MODEL);
    float q = 0.f;
#pragma unroll
    for (int j = 0; j < 8; ++j) { float d = v[j] - mean; q += d * d; }
#pragma unroll
    for (int off = 16; off > 0; off >>= 1) q += __shfl_xor(q, off, 32);
    const float rs = rsqrtf(q * (1.0f / D_MODEL) + EPSV);
    float* yr = y + (size_t)row * D_MODEL;
#pragma unroll
    for (int j = 0; j < 8; ++j) {
        const int dpos = lane + j * 32;
        yr[dpos] = (v[j] - mean) * rs * w[dpos] + b[dpos];
    }
}

// ---------------------------------------------------------------------------
// Causal depthwise conv (DC=4) over u = xz[:, 0:512] + SiLU.
// ---------------------------------------------------------------------------
__global__ void k_conv_silu(const float* __restrict__ xz,
                            const float* __restrict__ cw,
                            const float* __restrict__ cb,
                            float* __restrict__ uc)
{
    const int idx = blockIdx.x * blockDim.x + threadIdx.x;
    if (idx >= M_ROWS * D_INNER) return;
    const int d = idx & (D_INNER - 1);
    const int m = idx >> 9;
    const int n = m & (N_SEQ - 1);
    float acc = cb[d];
    const float* w = cw + d * D_CONV;
#pragma unroll
    for (int k = 0; k < D_CONV; ++k) {
        const int nn = n - (D_CONV - 1) + k;
        if (nn >= 0)
            acc += w[k] * xz[(size_t)(m - (D_CONV - 1) + k) * (2 * D_INNER) + d];
    }
    const float sg = 1.0f / (1.0f + __expf(-acc));
    uc[idx] = acc * sg;
}

// ---------------------------------------------------------------------------
// delta = softplus(dt @ dt_proj_w^T + dt_proj_b); dt = dbc[:, 0:16]. K=16 ->
// plain FMA is the right tool (too thin for WMMA).
// ---------------------------------------------------------------------------
__global__ void k_delta(const float* __restrict__ dbc,
                        const float* __restrict__ dtw,
                        const float* __restrict__ dtb,
                        float* __restrict__ delta)
{
    const int idx = blockIdx.x * blockDim.x + threadIdx.x;
    if (idx >= M_ROWS * D_INNER) return;
    const int d = idx & (D_INNER - 1);
    const int m = idx >> 9;
    const float* dt = dbc + (size_t)m * 48;
    const float* wr = dtw + d * D_TR;
    float acc = dtb[d];
#pragma unroll
    for (int r = 0; r < D_TR; ++r) acc = fmaf(dt[r], wr[r], acc);
    delta[idx] = (acc > 20.0f) ? acc : log1pf(__expf(acc));
}

// ---------------------------------------------------------------------------
// Selective scan fused with SiLU(z) gate. One lane per (batch, channel):
// 2048 lanes total, state h[16] + A[16] resident in VGPRs. Writes gated y
// in place over the delta buffer.
// ---------------------------------------------------------------------------
__global__ __launch_bounds__(256) void k_scan(
    float* delta_y,                    // in: delta  -> out: gated y
    const float* __restrict__ uc,
    const float* __restrict__ dbc,     // [m,48]: dt | B(16) | C(16)
    const float* __restrict__ xz,      // z at columns 512..1023
    const float* __restrict__ A_log,   // [512,16]
    const float* __restrict__ Dp)      // [512]
{
    const int gid = blockIdx.x * blockDim.x + threadIdx.x;
    if (gid >= BATCH * D_INNER) return;
    const int b = gid >> 9;
    const int d = gid & (D_INNER - 1);

    float A[D_STATE], h[D_STATE];
#pragma unroll
    for (int s = 0; s < D_STATE; ++s) {
        A[s] = -__expf(A_log[d * D_STATE + s]);
        h[s] = 0.0f;
    }
    const float Dpd = Dp[d];

    for (int n = 0; n < N_SEQ; ++n) {
        const size_t m  = (size_t)b * N_SEQ + n;
        const float dlt = delta_y[m * D_INNER + d];
        const float uu  = uc[m * D_INNER + d];
        const float* bc = dbc + m * 48 + 16;
        const float* cc = dbc + m * 48 + 32;
        const float du  = dlt * uu;
        float y = 0.0f;
#pragma unroll
        for (int s = 0; s < D_STATE; ++s) {
            h[s] = __expf(dlt * A[s]) * h[s] + du * bc[s];
            y = fmaf(h[s], cc[s], y);
        }
        const float z  = xz[m * (2 * D_INNER) + D_INNER + d];
        const float sz = z / (1.0f + __expf(-z));
        delta_y[m * D_INNER + d] = (y + uu * Dpd) * sz;
    }
}

// ---------------------------------------------------------------------------
// Host-side orchestration.
// ---------------------------------------------------------------------------
extern "C" void kernel_launch(void* const* d_in, const int* in_sizes, int n_in,
                              void* d_out, int out_size, void* d_ws, size_t ws_size,
                              hipStream_t stream)
{
    (void)in_sizes; (void)n_in; (void)out_size; (void)ws_size;

    const float* in_v    = (const float*)d_in[0];
    const float* in_a    = (const float*)d_in[1];
    const float* in_t    = (const float*)d_in[2];
    const float* mbnw    = (const float*)d_in[3];
    const float* mbnb    = (const float*)d_in[4];
    const float* inpw    = (const float*)d_in[5];
    const float* convw   = (const float*)d_in[6];
    const float* convb   = (const float*)d_in[7];
    const float* xpw     = (const float*)d_in[8];
    const float* dtw     = (const float*)d_in[9];
    const float* dtb     = (const float*)d_in[10];
    const float* Alog    = (const float*)d_in[11];
    const float* Dpp     = (const float*)d_in[12];
    const float* outw    = (const float*)d_in[13];
    const float* cplw    = (const float*)d_in[14];
    const float* coef    = (const float*)d_in[15];
    const float* fusew   = (const float*)d_in[16];
    const float* fuseb   = (const float*)d_in[17];
    const float* normw   = (const float*)d_in[18];
    const float* normb   = (const float*)d_in[19];

    const int M = M_ROWS;
    float* p   = (float*)d_ws;
    float* vb  = p; p += (size_t)M * 256;
    float* ab  = p; p += (size_t)M * 256;
    float* tb  = p; p += (size_t)M * 256;
    float* vin = p; p += (size_t)M * 256;
    float* ain = p; p += (size_t)M * 256;
    float* tin = p; p += (size_t)M * 256;
    float* xn  = p; p += (size_t)M * 256;
    float* xz  = p; p += (size_t)M * 1024;
    float* uc  = p; p += (size_t)M * 512;
    float* dbc = p; p += (size_t)M * 48;
    float* dlt = p; p += (size_t)M * 512;

    // Seed state buffers without mutating inputs (capture-legal async D2D).
    hipMemcpyAsync(vb, in_v, (size_t)M * 256 * sizeof(float), hipMemcpyDeviceToDevice, stream);
    hipMemcpyAsync(ab, in_a, (size_t)M * 256 * sizeof(float), hipMemcpyDeviceToDevice, stream);
    hipMemcpyAsync(tb, in_t, (size_t)M * 256 * sizeof(float), hipMemcpyDeviceToDevice, stream);

    auto gemm = [&](const float* X, const float* W, int ldw, const float* bias,
                    const float* resid, float* Y, int N, int K,
                    const float* alpha_p = nullptr, int alpha_idx = 0) {
        dim3 g((N + 127) / 128, (M + 63) / 64);
        k_gemm_wmma<<<g, 256, 0, stream>>>(X, W, bias, resid, Y, M, N, K, ldw,
                                           alpha_p, alpha_idx);
    };

    float* bufs_in[3] = {vin, ain, tin};
    float* bufs_st[3] = {vb, ab, tb};

    const int pw_grid = (M * D_INNER + 255) / 256;

    for (int i = 0; i < NLAYER; ++i) {
        // Couplings (all read the pre-update v/a/t):
        gemm(ab, cplw + (size_t)(i * 3 + 0) * 256 * 256, 256, nullptr, vb, vin, 256, 256, coef, i * 3 + 0);
        gemm(tb, cplw + (size_t)(i * 3 + 1) * 256 * 256, 256, nullptr, ab, ain, 256, 256, coef, i * 3 + 1);
        gemm(vb, cplw + (size_t)(i * 3 + 2) * 256 * 256, 256, nullptr, tb, tin, 256, 256, coef, i * 3 + 2);

        for (int j = 0; j < 3; ++j) {
            const int blk = 3 * i + j;
            const float* xin = bufs_in[j];
            float* xout = bufs_st[j];

            k_layernorm<<<M / 8, 256, 0, stream>>>(xin, mbnw + blk * 256, mbnb + blk * 256, xn, M);
            gemm(xn, inpw + (size_t)blk * 1024 * 256, 256, nullptr, nullptr, xz, 1024, 256);
            k_conv_silu<<<pw_grid, 256, 0, stream>>>(xz, convw + blk * 512 * 4, convb + blk * 512, uc);
            gemm(uc, xpw + (size_t)blk * 48 * 512, 512, nullptr, nullptr, dbc, 48, 512);
            k_delta<<<pw_grid, 256, 0, stream>>>(dbc, dtw + (size_t)blk * 512 * 16, dtb + blk * 512, dlt);
            k_scan<<<(BATCH * D_INNER + 255) / 256, 256, 0, stream>>>(
                dlt, uc, dbc, xz, Alog + (size_t)blk * 512 * 16, Dpp + blk * 512);
            // out = x_in + y_gated @ out_proj_w^T
            gemm(dlt, outw + (size_t)blk * 256 * 512, 512, nullptr, xin, xout, 256, 512);
        }
    }

    // Fused output projection: chain three K=256 chunks of fuse_w [256, 768].
    float* out = (float*)d_out;
    gemm(vb, fusew + 0,   768, fuseb,   nullptr, out, 256, 256);
    gemm(ab, fusew + 256, 768, nullptr, out,     out, 256, 256);
    gemm(tb, fusew + 512, 768, nullptr, out,     out, 256, 256);
    k_layernorm<<<M / 8, 256, 0, stream>>>(out, normw, normb, out, M);
}
